// GPT_34119220199746
// MI455X (gfx1250) — matrix operand order repaired and checked by reading, not compiled, not run on hardware
//
#include <hip/hip_runtime.h>

typedef __bf16 bf16;
typedef __attribute__((ext_vector_type(16))) __bf16 v16bf;
typedef __attribute__((ext_vector_type(8)))  __bf16 v8bf;
typedef __attribute__((ext_vector_type(8)))  float  v8f;
typedef __attribute__((ext_vector_type(4)))  unsigned int u32x4;
typedef __attribute__((ext_vector_type(4)))  int i32x4;
typedef __attribute__((ext_vector_type(8)))  int i32x8;

#define S_TOT   2048
#define T_CTX   1024
#define D_MODEL 768
#define N_HEAD  12
#define DHEAD   64
#define FF_DIM  3072
#define N_LAYER 12
#define VOCAB   50257

union FragU { v16bf v; v8bf h[2]; };

static __device__ __forceinline__ v16bf frag2(v8bf lo, v8bf hi) {
  FragU u; u.h[0] = lo; u.h[1] = hi; return u.v;
}

static __device__ __forceinline__ float redmax16(float v) {
  v = fmaxf(v, __shfl_xor(v, 1, 32));
  v = fmaxf(v, __shfl_xor(v, 2, 32));
  v = fmaxf(v, __shfl_xor(v, 4, 32));
  v = fmaxf(v, __shfl_xor(v, 8, 32));
  return v;
}
static __device__ __forceinline__ float redsum16(float v) {
  v += __shfl_xor(v, 1, 32);
  v += __shfl_xor(v, 2, 32);
  v += __shfl_xor(v, 4, 32);
  v += __shfl_xor(v, 8, 32);
  return v;
}

// ---- Tensor Data Mover: async 2D tile load global -> LDS with HW padding ----
// D# group0: [1:0]=count, [63:32]=lds_addr, [120:64]=global_addr, [127:126]=type(2)
// D# group1: [17:16]=data_size, [20]=pad_en, [24:22]=pad_interval, [31:25]=pad_amount,
//            [79:48]=tensor_dim0, [111:80]=tensor_dim1, [127:112]=tile_dim0,
//            [143:128]=tile_dim1, [207:160]=tensor_dim0_stride
static __device__ __forceinline__ void tdm_load_2d(
    unsigned lds_off, const void* g, unsigned tdim0, unsigned tdim1,
    unsigned tile0, unsigned tile1, unsigned long long stride0_elems,
    unsigned dsize_code, unsigned pad_interval, unsigned pad_amount)
{
  unsigned long long ga = (unsigned long long)(size_t)g;
  u32x4 g0;
  g0.x = 1u;                                            // one valid descriptor
  g0.y = lds_off;                                       // LDS byte address
  g0.z = (unsigned)ga;                                  // global addr lo
  g0.w = (unsigned)((ga >> 32) & 0x1FFFFFFull) | (2u << 30);  // hi + type=2
  unsigned long long q0 =
      ((unsigned long long)(dsize_code & 3u) << 16) |
      (1ull << 20) |                                    // pad_enable
      ((unsigned long long)(pad_interval & 7u) << 22) |
      ((unsigned long long)(pad_amount & 127u) << 25) |
      ((unsigned long long)(tdim0 & 0xFFFFu) << 48);
  unsigned long long q1 =
      (unsigned long long)(tdim0 >> 16) |
      ((unsigned long long)tdim1 << 16) |
      ((unsigned long long)(tile0 & 0xFFFFu) << 48);
  unsigned long long q2 =
      (unsigned long long)(tile1 & 0xFFFFu) |
      ((stride0_elems & 0xFFFFFFFFull) << 32);
  unsigned long long q3 = (stride0_elems >> 32) & 0xFFFFull;
  i32x8 g1;
  g1[0] = (int)(unsigned)q0; g1[1] = (int)(unsigned)(q0 >> 32);
  g1[2] = (int)(unsigned)q1; g1[3] = (int)(unsigned)(q1 >> 32);
  g1[4] = (int)(unsigned)q2; g1[5] = (int)(unsigned)(q2 >> 32);
  g1[6] = (int)(unsigned)q3; g1[7] = (int)(unsigned)(q3 >> 32);
  i32x4 z4 = {0, 0, 0, 0};
#if __clang_major__ >= 23
  i32x8 z8 = {0, 0, 0, 0, 0, 0, 0, 0};
  __builtin_amdgcn_tensor_load_to_lds(g0, g1, z4, z4, z8, 0);
#else
  __builtin_amdgcn_tensor_load_to_lds(g0, g1, z4, z4, 0);
#endif
}

// ---------------- embed: x = tok[ids] + pos ----------------
__global__ void __launch_bounds__(256)
embed_kernel(const int* __restrict__ ids, const float* __restrict__ tok,
             const float* __restrict__ pos, float* __restrict__ X)
{
  int gid = blockIdx.x * 256 + threadIdx.x;       // < 2048*768
  int d = gid % D_MODEL;
  int s = gid / D_MODEL;
  int t = s & (T_CTX - 1);
  X[gid] = tok[(size_t)ids[s] * D_MODEL + d] + pos[(size_t)t * D_MODEL + d];
}

// ---------------- layernorm (f32 in) -> bf16 out ----------------
__global__ void __launch_bounds__(256)
ln_kernel(const float* __restrict__ X, const float* __restrict__ G,
          const float* __restrict__ Bv, bf16* __restrict__ Out)
{
  __shared__ float red[256];
  int row = blockIdx.x, tid = threadIdx.x;
  const float* xr = X + (size_t)row * D_MODEL;
  float v0 = xr[tid], v1 = xr[tid + 256], v2 = xr[tid + 512];
  red[tid] = v0 + v1 + v2;
  __syncthreads();
  for (int o = 128; o > 0; o >>= 1) { if (tid < o) red[tid] += red[tid + o]; __syncthreads(); }
  float mean = red[0] * (1.0f / (float)D_MODEL);
  __syncthreads();
  float d0 = v0 - mean, d1 = v1 - mean, d2 = v2 - mean;
  red[tid] = d0 * d0 + d1 * d1 + d2 * d2;
  __syncthreads();
  for (int o = 128; o > 0; o >>= 1) { if (tid < o) red[tid] += red[tid + o]; __syncthreads(); }
  float rstd = rsqrtf(red[0] * (1.0f / (float)D_MODEL) + 1e-5f);
  bf16* orow = Out + (size_t)row * D_MODEL;
  orow[tid]       = (bf16)(d0 * rstd * G[tid]       + Bv[tid]);
  orow[tid + 256] = (bf16)(d1 * rstd * G[tid + 256] + Bv[tid + 256]);
  orow[tid + 512] = (bf16)(d2 * rstd * G[tid + 512] + Bv[tid + 512]);
}

// ---------------- tiled WMMA GEMM with TDM double-buffered staging ----------------
//   C[M,N] = A_bf16[M,K] @ W_f32[N,K]^T      (M = 2048 fixed)
// EPI: 0 = f32 store, 1 = +residual f32 store, 2 = exact GELU -> bf16, 3 = bf16 store
template<int EPI>
__global__ void __launch_bounds__(256)
gemm_bf16(const bf16* __restrict__ A, const float* __restrict__ W,
          float* __restrict__ Cf, bf16* __restrict__ Cb,
          const float* __restrict__ R, int N, int K)
{
  // TDM pads each 64B A-row by 16B -> stride 40 bf16; each 128B B-row by 16B -> stride 36 f32
  __shared__ __align__(16) bf16  As[2][128][40];
  __shared__ __align__(16) float Bsf[2][128][36];
  const int tid  = threadIdx.x;
  const int lane = tid & 31;
  const int l15  = lane & 15;
  const int hsel = (lane >> 4) & 1;
  const int w    = tid >> 5;
  const int wm   = w >> 1;              // 0..3 -> 32-row strip
  const int wn   = w & 1;               // 0..1 -> 64-col strip
  const int m0   = blockIdx.y * 128;
  const int n0   = blockIdx.x * 128;

  auto issueA = [&](int buf, int k0) {
    tdm_load_2d((unsigned)(size_t)&As[buf][0][0],
                A + (size_t)m0 * K + k0,
                (unsigned)(K - k0), (unsigned)(S_TOT - m0),
                32u, 128u, (unsigned long long)K,
                /*dsize 2B*/1u, /*pad every 16 DW*/3u, /*pad 4 DW*/3u);
  };
  auto issueB = [&](int buf, int k0) {
    tdm_load_2d((unsigned)(size_t)&Bsf[buf][0][0],
                W + (size_t)n0 * K + k0,
                (unsigned)(K - k0), (unsigned)(N - n0),   // OOB rows -> zero fill
                32u, 128u, (unsigned long long)K,
                /*dsize 4B*/2u, /*pad every 32 DW*/4u, /*pad 4 DW*/3u);
  };

  v8f acc[2][4];
#pragma unroll
  for (int i = 0; i < 2; i++)
#pragma unroll
    for (int j = 0; j < 4; j++) acc[i][j] = {};

  if (w == 0) {
    issueA(0, 0); issueB(0, 0);
    __builtin_amdgcn_s_wait_tensorcnt(0);
  }
  __syncthreads();

  const int nsteps = K >> 5;
  for (int s = 0; s < nsteps; ++s) {
    const int cur = s & 1;
    if (w == 0 && s + 1 < nsteps) {           // kick off next tile DMA, overlaps math
      issueA(cur ^ 1, (s + 1) << 5);
      issueB(cur ^ 1, (s + 1) << 5);
    }

    v16bf af[2];
#pragma unroll
    for (int i = 0; i < 2; i++) {
      // A layout: lane=row; half-lanes split K as {0..7,16..23} / {8..15,24..31}
      const bf16* ap = &As[cur][wm * 32 + i * 16 + l15][hsel * 8];
      af[i] = frag2(*(const v8bf*)ap, *(const v8bf*)(ap + 16));
    }
#pragma unroll
    for (int j = 0; j < 4; j++) {
      // B layout: lane=col; half-lanes split K as {0..7,16..23} / {8..15,24..31} (cvt f32->bf16)
      const float* bp = &Bsf[cur][wn * 64 + j * 16 + l15][hsel * 8];
      v16bf bfrag;
#pragma unroll
      for (int e = 0; e < 16; e++) bfrag[e] = (bf16)bp[e + (e & 8)];
#pragma unroll
      for (int i = 0; i < 2; i++)
        acc[i][j] = __builtin_amdgcn_wmma_f32_16x16x32_bf16(
            false, af[i], false, bfrag, (short)0, acc[i][j], false, false);
    }

    if (w == 0 && s + 1 < nsteps) __builtin_amdgcn_s_wait_tensorcnt(0);
    __syncthreads();                          // publish next buffer / retire current
  }

  // epilogue: C layout lane=col, vgpr r = row r (+8 for upper half-lanes)
#pragma unroll
  for (int i = 0; i < 2; i++)
#pragma unroll
    for (int j = 0; j < 4; j++)
#pragma unroll
      for (int r = 0; r < 8; r++) {
        int row = m0 + wm * 32 + i * 16 + r + hsel * 8;
        int col = n0 + wn * 64 + j * 16 + l15;
        if (col < N) {
          size_t idx = (size_t)row * N + col;
          float v = acc[i][j][r];
          if (EPI == 0) {
            Cf[idx] = v;
          } else if (EPI == 1) {
            Cf[idx] = v + R[idx];
          } else if (EPI == 2) {
            Cb[idx] = (bf16)(0.5f * v * (1.0f + erff(v * 0.70710678118f)));
          } else {
            Cb[idx] = (bf16)v;
          }
        }
      }
}

// ---------------- split qkv -> q[b,h,s,d], k[b,h,s,d], vT[b,h,d,s] (all bf16) ----------------
__global__ void __launch_bounds__(256)
qkv_split_kernel(const bf16* __restrict__ QKV, bf16* __restrict__ Q,
                 bf16* __restrict__ Kk, bf16* __restrict__ VT)
{
  int gid = blockIdx.x * 256 + threadIdx.x;   // < 2*12*1024*64
  int d  = gid & 63;
  int s  = (gid >> 6) & (T_CTX - 1);
  int bh = gid >> 16;
  int b  = bh / N_HEAD, hh = bh % N_HEAD;
  size_t src = (size_t)(b * T_CTX + s) * (3 * D_MODEL) + hh * DHEAD + d;
  Q[gid]  = QKV[src];
  Kk[gid] = QKV[src + D_MODEL];
  VT[((size_t)bh * DHEAD + d) * T_CTX + s] = QKV[src + 2 * D_MODEL];
}

// ---------------- flash attention: per (b, head, 64-query tile), 4 waves x 16 rows ----------------
__global__ void __launch_bounds__(128)
attn_kernel(const bf16* __restrict__ Q, const bf16* __restrict__ Kk,
            const bf16* __restrict__ VT, bf16* __restrict__ O)
{
  __shared__ __align__(16) bf16 pw[4][16][72];   // per-wave 16x64 prob tile, padded
  const int tid  = threadIdx.x;
  const int lane = tid & 31;
  const int l15  = lane & 15;
  const int hsel = (lane >> 4) & 1;
  const int w    = tid >> 5;            // wave 0..3
  const int qb   = blockIdx.x;          // 0..15
  const int hd   = blockIdx.y;
  const int b    = blockIdx.z;
  const int bh   = b * N_HEAD + hd;
  const int q0   = qb * 64 + w * 16;    // this wave's 16 query rows

  // Q fragments (K = DHEAD = 64 -> two 16x32 frags), A layout
  const bf16* qrow = Q + ((size_t)bh * T_CTX + q0 + l15) * DHEAD;
  v16bf qf[2];
#pragma unroll
  for (int f = 0; f < 2; f++) {
    int ko = f * 32 + hsel * 8;
    qf[f] = frag2(*(const v8bf*)(qrow + ko), *(const v8bf*)(qrow + ko + 16));
  }

  v8f oacc[4];
#pragma unroll
  for (int j = 0; j < 4; j++) oacc[j] = {};
  float m_r[8], l_r[8];
#pragma unroll
  for (int r = 0; r < 8; r++) { m_r[r] = -1e30f; l_r[r] = 0.f; }
  const float scale = 0.125f;   // 1/sqrt(64)

  for (int kc = 0; kc <= q0 + 15; kc += 64) {
    // scores S = Q K^T (4 key tiles of 16)
    v8f s[4];
#pragma unroll
    for (int j = 0; j < 4; j++) {
      s[j] = {};
      const bf16* krow = Kk + ((size_t)bh * T_CTX + kc + j * 16 + l15) * DHEAD;
#pragma unroll
      for (int f = 0; f < 2; f++) {
        int ko = f * 32 + hsel * 8;   // B layout: half-lanes split K as {0..7,16..23} / {8..15,24..31}
        v16bf kf = frag2(*(const v8bf*)(krow + ko), *(const v8bf*)(krow + ko + 16));
        s[j] = __builtin_amdgcn_wmma_f32_16x16x32_bf16(
            false, qf[f], false, kf, (short)0, s[j], false, false);
      }
    }
    // scale + causal mask (C layout: row = q0 + r + hsel*8, col = kc + j*16 + l15)
#pragma unroll
    for (int j = 0; j < 4; j++) {
      int col = kc + j * 16 + l15;
#pragma unroll
      for (int r = 0; r < 8; r++) {
        int row = q0 + r + hsel * 8;
        float v = s[j][r] * scale;
        s[j][r] = (col > row) ? -1e30f : v;
      }
    }
    // online softmax per row (16-lane shuffle reductions)
#pragma unroll
    for (int r = 0; r < 8; r++) {
      float cm = fmaxf(fmaxf(s[0][r], s[1][r]), fmaxf(s[2][r], s[3][r]));
      cm = redmax16(cm);
      float mn = fmaxf(m_r[r], cm);
      float rs = __expf(m_r[r] - mn);
      float psum = 0.f;
#pragma unroll
      for (int j = 0; j < 4; j++) {
        float pv = __expf(s[j][r] - mn);
        s[j][r] = pv;
        psum += pv;
      }
      psum = redsum16(psum);
      l_r[r] = l_r[r] * rs + psum;
      m_r[r] = mn;
#pragma unroll
      for (int j = 0; j < 4; j++) oacc[j][r] *= rs;
    }
    // stage probabilities to per-wave LDS (C layout -> row-major 16x64 bf16)
#pragma unroll
    for (int j = 0; j < 4; j++)
#pragma unroll
      for (int r = 0; r < 8; r++)
        pw[w][r + hsel * 8][j * 16 + l15] = (bf16)s[j][r];
    asm volatile("s_wait_dscnt 0" ::: "memory");
    // reload probs in A layout, accumulate O += P @ V (V^T is [d, keys], key-contiguous)
    v16bf pf[2];
#pragma unroll
    for (int f = 0; f < 2; f++) {
      const bf16* pp = &pw[w][l15][f * 32 + hsel * 8];
      pf[f] = frag2(*(const v8bf*)pp, *(const v8bf*)(pp + 16));
    }
#pragma unroll
    for (int j = 0; j < 4; j++) {
      const bf16* vrow = VT + ((size_t)bh * DHEAD + j * 16 + l15) * T_CTX + kc;
#pragma unroll
      for (int f = 0; f < 2; f++) {
        int ko = f * 32 + hsel * 8;
        v16bf vf = frag2(*(const v8bf*)(vrow + ko), *(const v8bf*)(vrow + ko + 16));
        oacc[j] = __builtin_amdgcn_wmma_f32_16x16x32_bf16(
            false, pf[f], false, vf, (short)0, oacc[j], false, false);
      }
    }
  }

  // normalize and write bf16 output [S_TOT, D_MODEL]
#pragma unroll
  for (int j = 0; j < 4; j++)
#pragma unroll
    for (int r = 0; r < 8; r++) {
      int row = b * T_CTX + q0 + r + hsel * 8;
      int col = hd * DHEAD + j * 16 + l15;
      O[(size_t)row * D_MODEL + col] = (bf16)(oacc[j][r] / l_r[r]);
    }
}

// ---------------- host orchestration ----------------
extern "C" void kernel_launch(void* const* d_in, const int* in_sizes, int n_in,
                              void* d_out, int out_size, void* d_ws, size_t ws_size,
                              hipStream_t stream)
{
  (void)in_sizes; (void)n_in; (void)out_size; (void)ws_size;
  const int*   ids  = (const int*)  d_in[0];
  const float* tok  = (const float*)d_in[1];
  const float* pos  = (const float*)d_in[2];
  const float* ln1g = (const float*)d_in[3];
  const float* ln1b = (const float*)d_in[4];
  const float* qkvw = (const float*)d_in[5];
  const float* outw = (const float*)d_in[6];
  const float* ln2g = (const float*)d_in[7];
  const float* ln2b = (const float*)d_in[8];
  const float* ff1w = (const float*)d_in[9];
  const float* ff2w = (const float*)d_in[10];
  const float* lnfg = (const float*)d_in[11];
  const float* lnfb = (const float*)d_in[12];

  char* p = (char*)d_ws;
  float* x   = (float*)p; p += (size_t)S_TOT * D_MODEL * 4;       // residual stream f32
  bf16* h    = (bf16*)p;  p += (size_t)S_TOT * D_MODEL * 2;       // LN output bf16
  bf16* qkvb = (bf16*)p;  p += (size_t)S_TOT * 3 * D_MODEL * 2;   // qkv bf16
  bf16* qb   = (bf16*)p;  p += (size_t)S_TOT * D_MODEL * 2;
  bf16* kb   = (bf16*)p;  p += (size_t)S_TOT * D_MODEL * 2;
  bf16* vtb  = (bf16*)p;  p += (size_t)S_TOT * D_MODEL * 2;       // V transposed
  bf16* obf  = (bf16*)p;  p += (size_t)S_TOT * D_MODEL * 2;       // attention out bf16
  bf16* ffa  = (bf16*)p;  p += (size_t)S_TOT * FF_DIM * 2;        // gelu(ff1) bf16

  embed_kernel<<<S_TOT * D_MODEL / 256, 256, 0, stream>>>(ids, tok, pos, x);

  for (int l = 0; l < N_LAYER; ++l) {
    ln_kernel<<<S_TOT, 256, 0, stream>>>(x, ln1g + l * D_MODEL, ln1b + l * D_MODEL, h);
    gemm_bf16<3><<<dim3(18, 16), 256, 0, stream>>>(
        h, qkvw + (size_t)l * 3 * D_MODEL * D_MODEL, nullptr, qkvb, nullptr,
        3 * D_MODEL, D_MODEL);
    qkv_split_kernel<<<S_TOT * D_MODEL / 256, 256, 0, stream>>>(qkvb, qb, kb, vtb);
    attn_kernel<<<dim3(16, N_HEAD, 2), 128, 0, stream>>>(qb, kb, vtb, obf);
    gemm_bf16<1><<<dim3(6, 16), 256, 0, stream>>>(
        obf, outw + (size_t)l * D_MODEL * D_MODEL, x, nullptr, x, D_MODEL, D_MODEL);
    ln_kernel<<<S_TOT, 256, 0, stream>>>(x, ln2g + l * D_MODEL, ln2b + l * D_MODEL, h);
    gemm_bf16<2><<<dim3(24, 16), 256, 0, stream>>>(
        h, ff1w + (size_t)l * FF_DIM * D_MODEL, nullptr, ffa, nullptr, FF_DIM, D_MODEL);
    gemm_bf16<1><<<dim3(6, 16), 256, 0, stream>>>(
        ffa, ff2w + (size_t)l * D_MODEL * FF_DIM, x, nullptr, x, D_MODEL, FF_DIM);
  }

  ln_kernel<<<S_TOT, 256, 0, stream>>>(x, lnfg, lnfb, h);
  gemm_bf16<0><<<dim3((VOCAB + 127) / 128, 16), 256, 0, stream>>>(
      h, tok, (float*)d_out, nullptr, nullptr, VOCAB, D_MODEL);
}
